// CPAMDec_Mix_55439437857400
// MI455X (gfx1250) — matrix operand order repaired and checked by reading, not compiled, not run on hardware
//
#include <hip/hip_runtime.h>

// Problem constants (match reference)
#define NPIX 4096   // W*H
#define CCH  512    // C
#define KK   128    // K
#define BB   16     // B

typedef __attribute__((ext_vector_type(16))) __bf16 v16bf;
typedef __attribute__((ext_vector_type(8)))  __bf16 v8bf;
typedef __attribute__((ext_vector_type(8)))  float  v8f;

__device__ __forceinline__ v8f wmma_bf16(v16bf a, v16bf b, v8f c) {
  // D = A(16x32 bf16) * B(32x16 bf16) + C(16x16 f32)
  return __builtin_amdgcn_wmma_f32_16x16x32_bf16(false, a, false, b, (short)0, c,
                                                 false, false);
}

// ---- fragments from fp32 sources (producer kernels only; one-time convert) ----
// A-fragment (16x32 bf16): lanes 0-15 hold K=0..15, lanes 16-31 K=16..31.
__device__ __forceinline__ v16bf a_frag_f32(const float* __restrict__ rowp, int k0, int hi) {
  const float* p0 = rowp + k0 + hi * 16;
  const float* p1 = p0 + 8;
  v16bf f;
#pragma unroll
  for (int j = 0; j < 8; ++j) { f[j] = (__bf16)p0[j]; f[j + 8] = (__bf16)p1[j]; }
  return f;
}
// B-fragment (32x16 bf16) from "B-transposed" row-major [ncol][k] source;
// lanes 0-15 hold K=0..15, lanes 16-31 K=16..31.
__device__ __forceinline__ v16bf b_frag_f32(const float* __restrict__ colp, int k0, int hi) {
  const float* p0 = colp + k0 + hi * 16;
  v16bf f;
#pragma unroll
  for (int j = 0; j < 16; ++j) f[j] = (__bf16)p0[j];
  return f;
}

// ---- fragments from packed bf16 sources (global workspace or LDS panels) ----
// Pure b128 loads, no conversion VALU. Works for generic pointers (LDS or global).
__device__ __forceinline__ v16bf a_frag_bf(const __bf16* rowp, int k0, int hi) {
  v8bf lo = *(const v8bf*)(rowp + k0 + hi * 16);
  v8bf hv = *(const v8bf*)(rowp + k0 + hi * 16 + 8);
  return __builtin_shufflevector(lo, hv, 0, 1, 2, 3, 4, 5, 6, 7, 8, 9, 10, 11, 12, 13, 14, 15);
}
__device__ __forceinline__ v16bf b_frag_bf(const __bf16* colp, int k0, int hi) {
  // 16 consecutive bf16 = 32B, 32B-aligned for all our strides.
  return *(const v16bf*)(colp + k0 + hi * 16);
}

// ---------------------------------------------------------------------------
// Stage 1: K1,K2 = y @ wk^T + bk        -> bf16 row-major [128 x 256] per batch
//          VT1,VT2 = (y @ wv^T + bv)^T  -> bf16 row-major [512 x 128] per batch
// One 16x16 tile per wave, C=512 inner (16 WMMA steps), fp32 accumulate,
// bf16 store. Inputs (y: 8MB, weights: 3MB) are L2-resident on the 192MB L2,
// so tile-per-wave load amplification never reaches HBM.
// ---------------------------------------------------------------------------
__global__ __launch_bounds__(256) void kv_proj_kernel(
    const float* __restrict__ y1, const float* __restrict__ y2,
    const float* __restrict__ wk1, const float* __restrict__ bk1,
    const float* __restrict__ wk2, const float* __restrict__ bk2,
    const float* __restrict__ wv1, const float* __restrict__ bv1,
    const float* __restrict__ wv2, const float* __restrict__ bv2,
    __bf16* __restrict__ K1, __bf16* __restrict__ K2,
    __bf16* __restrict__ VT1, __bf16* __restrict__ VT2)
{
  const int lane = threadIdx.x & 31;
  const int ln = lane & 15, hi = lane >> 4;
  const int wave = threadIdx.x >> 5;
  int t = blockIdx.x * 8 + wave;         // 16 batches * 768 tiles
  int b = t / 768;
  int r = t - b * 768;

  const float *A, *BT, *bias;
  __bf16* D;
  int ldd, rowt, colt, rowbias;
  if (r < 256) {                          // K projections: [128 x 256]
    int s  = r >> 7;
    int rr = r & 127;                     // 8 x 16 tiles
    rowt = rr >> 4; colt = rr & 15;
    A    = (s ? y2 : y1) + (size_t)b * KK * CCH;     // rows k, inner c
    BT   = s ? wk2 : wk1;                            // rows d, inner c
    bias = s ? bk2 : bk1;
    D    = (s ? K2 : K1) + (size_t)b * KK * 256;
    ldd = 256; rowbias = 0;               // bias per output column d
  } else {                                // V projections, transposed: [512 x 128]
    int r2 = r - 256;
    int s  = r2 >> 8;
    int rr = r2 & 255;                    // 32 x 8 tiles
    rowt = rr >> 3; colt = rr & 7;
    A    = s ? wv2 : wv1;                            // rows c, inner c'
    BT   = (s ? y2 : y1) + (size_t)b * KK * CCH;     // rows k, inner c'
    bias = s ? bv2 : bv1;
    D    = (s ? VT2 : VT1) + (size_t)b * CCH * KK;
    ldd = KK; rowbias = 1;                // bias per output row c
  }

  const float* arow = A + (size_t)(rowt * 16 + ln) * CCH;
  const float* bcol = BT + (size_t)(colt * 16 + ln) * CCH;
  v8f acc = {};
#pragma unroll
  for (int k0 = 0; k0 < CCH; k0 += 32) {
    v16bf af = a_frag_f32(arow, k0, hi);
    v16bf bf = b_frag_f32(bcol, k0, hi);
    acc = wmma_bf16(af, bf, acc);
  }
  float bcolv = bias[colt * 16 + ln];
#pragma unroll
  for (int i = 0; i < 8; ++i) {
    int m = i + hi * 8;                   // D layout: VGPR i -> rows i / i+8
    float bvv = rowbias ? bias[rowt * 16 + m] : bcolv;
    D[(size_t)(rowt * 16 + m) * ldd + colt * 16 + ln] = (__bf16)(acc[i] + bvv);
  }
}

// ---------------------------------------------------------------------------
// Stage 2: Q[b,n,0:128]=x1-proj, Q[b,n,128:256]=x2-proj, bf16 row-major [N x 256].
// x is [C,N] (N contiguous) so x tiles are transposed into an LDS bf16 A-panel;
// the weight panel is staged straight ([d][c] row-major).
// Workgroup = 128 n-rows x 128 d-cols, 8 waves, one wave per n-tile strip.
// ---------------------------------------------------------------------------
__global__ __launch_bounds__(256) void q_proj_kernel(
    const float* __restrict__ x1, const float* __restrict__ x2,
    const float* __restrict__ wq1, const float* __restrict__ bq1,
    const float* __restrict__ wq2, const float* __restrict__ bq2,
    __bf16* __restrict__ Q)
{
  __shared__ __bf16 Apan[128][48];   // [n][c] panel, stride 48 (16B-aligned rows, de-pow2)
  __shared__ __bf16 Bpan[128][32];   // [d][c] panel (64B rows -> 32B-aligned B-frags)
  const int nblk = blockIdx.x, s = blockIdx.y, b = blockIdx.z;
  const float* x  = s ? x2 : x1;
  const float* wq = s ? wq2 : wq1;
  const float* bq = s ? bq2 : bq1;
  const int lane = threadIdx.x & 31, wave = threadIdx.x >> 5;
  const int ln = lane & 15, hi = lane >> 4;
  const int n0 = nblk * 128;

  v8f acc[8] = {};
  for (int c0 = 0; c0 < CCH; c0 += 32) {
    {  // stage x[b, c0..c0+32, n0..n0+128] transposed -> Apan[n][c]
      int cl = threadIdx.x >> 3;               // 0..31
      int no = (threadIdx.x & 7) * 16;         // 0..112
      const float* src = x + ((size_t)b * CCH + (c0 + cl)) * NPIX + n0 + no;
#pragma unroll
      for (int j = 0; j < 16; ++j) Apan[no + j][cl] = (__bf16)src[j];
    }
    {  // stage wq[0..128, c0..c0+32] -> Bpan[d][c] (already row-major)
      int d  = threadIdx.x >> 1;               // 0..127
      int co = (threadIdx.x & 1) * 16;
      const float* src = wq + (size_t)d * CCH + c0 + co;
#pragma unroll
      for (int j = 0; j < 16; ++j) Bpan[d][co + j] = (__bf16)src[j];
    }
    __syncthreads();
    v16bf af = a_frag_bf(&Apan[wave * 16 + ln][0], 0, hi);
#pragma unroll
    for (int dt = 0; dt < 8; ++dt) {
      v16bf bf = b_frag_bf(&Bpan[dt * 16 + ln][0], 0, hi);
      acc[dt] = wmma_bf16(af, bf, acc[dt]);
    }
    __syncthreads();
  }
#pragma unroll
  for (int dt = 0; dt < 8; ++dt) {
    float bias = bq[dt * 16 + ln];
#pragma unroll
    for (int i = 0; i < 8; ++i) {
      int m = i + hi * 8;
      int n = n0 + wave * 16 + m;
      int d = s * 128 + dt * 16 + ln;
      Q[((size_t)b * NPIX + n) * 256 + d] = (__bf16)(acc[dt][i] + bias);
    }
  }
}

// ---------------------------------------------------------------------------
// Stage 3 (fused): e1=Q K1^T, e2=Q K2^T, A=softmax(|e1-e2|),
// O_s = A V_s, out_s = scale*O_s + x_s.
// All GEMM operands arrive as packed bf16 -> fragment loads are raw
// global_load_b128 / ds_load_b128 with zero conversion VALU in the hot loops.
// Per-batch K/V set (0.4 MB bf16) is L2-resident; LDS only for layout changes.
// ---------------------------------------------------------------------------
__global__ __launch_bounds__(256) void attn_kernel(
    const __bf16* __restrict__ Q,  const __bf16* __restrict__ K1,
    const __bf16* __restrict__ K2, const __bf16* __restrict__ VT1,
    const __bf16* __restrict__ VT2,
    const float* __restrict__ x1, const float* __restrict__ x2,
    const float* __restrict__ scale_p, float* __restrict__ out)
{
  __shared__ __bf16 attnP[8][16][128];   // per-wave attention tile, 32 KB
  __shared__ float  tbuf[8][16][17];     // per-wave 16x16 transpose buffer, 8.5 KB
  const int nblk = blockIdx.x, b = blockIdx.y;
  const int lane = threadIdx.x & 31, wave = threadIdx.x >> 5;
  const int ln = lane & 15, hi = lane >> 4;
  const int nbase = nblk * 128 + wave * 16;

  // Preload this wave's Q fragments: 16 rows x 256 d = 8 A-fragments,
  // reused across 16 e-GEMM passes (8 kt-tiles x 2 streams).
  const __bf16* qrow = Q + ((size_t)b * NPIX + nbase + ln) * 256;
  v16bf qf[8];
#pragma unroll
  for (int j = 0; j < 8; ++j) qf[j] = a_frag_bf(qrow, j * 32, hi);

  // E = |Q K1^T - Q K2^T|, kept in WMMA D layout (8 col-tiles of 16).
  const __bf16* K1b = K1 + (size_t)b * KK * 256;
  const __bf16* K2b = K2 + (size_t)b * KK * 256;
  v8f E[8];
#pragma unroll
  for (int kt = 0; kt < 8; ++kt) {
    v8f a1 = {}, a2 = {};
    const __bf16* k1c = K1b + (size_t)(kt * 16 + ln) * 256;
    const __bf16* k2c = K2b + (size_t)(kt * 16 + ln) * 256;
#pragma unroll
    for (int j = 0; j < 8; ++j) {
      a1 = wmma_bf16(qf[j], b_frag_bf(k1c, j * 32, hi), a1);
      a2 = wmma_bf16(qf[j], b_frag_bf(k2c, j * 32, hi), a2);
    }
#pragma unroll
    for (int i = 0; i < 8; ++i) E[kt][i] = fabsf(a1[i] - a2[i]);
  }

  // Row softmax over k=128. A row's values sit at fixed VGPR slot i across the
  // 8 kt-sets and one 16-lane half of the wave: reduce kt-sets in registers,
  // then shuffle-xor within each 16-lane group (wave32 halves hold disjoint
  // rows, so masks 1..8 never mix rows).
  float mx[8], sm[8];
#pragma unroll
  for (int i = 0; i < 8; ++i) {
    float m = E[0][i];
#pragma unroll
    for (int kt = 1; kt < 8; ++kt) m = fmaxf(m, E[kt][i]);
    mx[i] = m;
  }
#pragma unroll
  for (int off = 1; off < 16; off <<= 1) {
#pragma unroll
    for (int i = 0; i < 8; ++i) mx[i] = fmaxf(mx[i], __shfl_xor(mx[i], off, 32));
  }
#pragma unroll
  for (int i = 0; i < 8; ++i) sm[i] = 0.f;
#pragma unroll
  for (int kt = 0; kt < 8; ++kt) {
#pragma unroll
    for (int i = 0; i < 8; ++i) {
      float p = __expf(E[kt][i] - mx[i]);
      E[kt][i] = p;
      sm[i] += p;
    }
  }
#pragma unroll
  for (int off = 1; off < 16; off <<= 1) {
#pragma unroll
    for (int i = 0; i < 8; ++i) sm[i] += __shfl_xor(sm[i], off, 32);
  }

  // Normalized attention -> LDS row-major [n][k] (bf16) to form A-fragments.
#pragma unroll
  for (int kt = 0; kt < 8; ++kt) {
#pragma unroll
    for (int i = 0; i < 8; ++i)
      attnP[wave][i + hi * 8][kt * 16 + ln] = (__bf16)(E[kt][i] / sm[i]);
  }
  __syncthreads();

  v16bf af[4];
#pragma unroll
  for (int j = 0; j < 4; ++j) af[j] = a_frag_bf(&attnP[wave][ln][0], j * 32, hi);

  // O = attn * V (VT stored [c][k] bf16 so B-fragments are contiguous rows),
  // then scale + residual, stores coalesced along n via LDS transpose.
  const float sc = scale_p[0];
#pragma unroll
  for (int s = 0; s < 2; ++s) {
    const __bf16* VT = (s ? VT2 : VT1) + (size_t)b * CCH * KK;
    const float* x  = s ? x2 : x1;
    float* o = out + (size_t)s * BB * CCH * NPIX;
    for (int ct = 0; ct < 32; ++ct) {
      const __bf16* vcol = VT + (size_t)(ct * 16 + ln) * KK;
      if (ct < 31) __builtin_prefetch(vcol + 16 * KK, 0, 3);  // next V panel -> global_prefetch
      v8f acc = {};
#pragma unroll
      for (int j = 0; j < 4; ++j) acc = wmma_bf16(af[j], b_frag_bf(vcol, j * 32, hi), acc);
      // D layout (lane=c, VGPR=n) -> tbuf[c][n]; DS ops are in-order per wave.
#pragma unroll
      for (int i = 0; i < 8; ++i) tbuf[wave][ln][i + hi * 8] = acc[i];
#pragma unroll
      for (int it = 0; it < 8; ++it) {
        int cr = it * 2 + hi;
        int cg = ct * 16 + cr;
        size_t idx = ((size_t)b * CCH + cg) * NPIX + (size_t)nblk * 128 + wave * 16 + ln;
        o[idx] = sc * tbuf[wave][cr][ln] + x[idx];   // 64B per half-wave
      }
    }
  }
}

// ---------------------------------------------------------------------------
extern "C" void kernel_launch(void* const* d_in, const int* in_sizes, int n_in,
                              void* d_out, int out_size, void* d_ws, size_t ws_size,
                              hipStream_t stream) {
  const float* x1    = (const float*)d_in[0];
  const float* y1    = (const float*)d_in[1];
  const float* x2    = (const float*)d_in[2];
  const float* y2    = (const float*)d_in[3];
  const float* wq1   = (const float*)d_in[4];
  const float* bq1   = (const float*)d_in[5];
  const float* wq2   = (const float*)d_in[6];
  const float* bq2   = (const float*)d_in[7];
  const float* wk1   = (const float*)d_in[8];
  const float* bk1   = (const float*)d_in[9];
  const float* wk2   = (const float*)d_in[10];
  const float* bk2   = (const float*)d_in[11];
  const float* wv1   = (const float*)d_in[12];
  const float* bv1   = (const float*)d_in[13];
  const float* wv2   = (const float*)d_in[14];
  const float* bv2   = (const float*)d_in[15];
  const float* scale = (const float*)d_in[16];
  float* out = (float*)d_out;

  // Workspace layout (bf16): Q | K1 | K2 | VT1 | VT2  (~40 MB total)
  __bf16* Qw  = (__bf16*)d_ws;
  __bf16* K1w = Qw  + (size_t)BB * NPIX * 256;
  __bf16* K2w = K1w + (size_t)BB * KK * 256;
  __bf16* V1w = K2w + (size_t)BB * KK * 256;
  __bf16* V2w = V1w + (size_t)BB * CCH * KK;

  kv_proj_kernel<<<dim3((BB * 768) / 8), 256, 0, stream>>>(
      y1, y2, wk1, bk1, wk2, bk2, wv1, bv1, wv2, bv2, K1w, K2w, V1w, V2w);
  q_proj_kernel<<<dim3(NPIX / 128, 2, BB), 256, 0, stream>>>(
      x1, x2, wq1, bq1, wq2, bq2, Qw);
  attn_kernel<<<dim3(NPIX / 128, BB), 256, 0, stream>>>(
      Qw, K1w, K2w, V1w, V2w, x1, x2, scale, out);

  (void)in_sizes; (void)n_in; (void)out_size; (void)ws_size;
}
